// CapsuleNetwork_35708358099457
// MI455X (gfx1250) — matrix operand order repaired and checked by reading, not compiled, not run on hardware
//
#include <hip/hip_runtime.h>
#include <hip/hip_bf16.h>
#include <math.h>

typedef __attribute__((ext_vector_type(16))) _Float16 v16h;
typedef __attribute__((ext_vector_type(8)))  _Float16 v8h;
typedef __attribute__((ext_vector_type(8)))  float    v8f;
typedef __attribute__((ext_vector_type(4)))  int      v4i;

// ---------------------------------------------------------------------------
// Async global->LDS copy (gfx1250 ASYNCcnt path) with portable fallback.
// Builtin signature (from probe diagnostic): (v4i AS1*, v4i AS3*, Ii, Ii).
// ---------------------------------------------------------------------------
#if __has_builtin(__builtin_amdgcn_global_load_async_to_lds_b128)
#define ASYNC_LDS 1
__device__ __forceinline__ void async_copy16(void* lds, const void* g) {
  __builtin_amdgcn_global_load_async_to_lds_b128(
      (__attribute__((address_space(1))) v4i*)(void*)(g),
      (__attribute__((address_space(3))) v4i*)(lds), 0, 0);
}
__device__ __forceinline__ void async_wait() {
#if __has_builtin(__builtin_amdgcn_s_wait_asynccnt)
  __builtin_amdgcn_s_wait_asynccnt(0);
#else
  asm volatile("s_wait_asynccnt 0x0" ::: "memory");
#endif
}
#else
#define ASYNC_LDS 0
__device__ __forceinline__ void async_copy16(void* lds, const void* g) {
  *(v8h*)lds = *(const v8h*)g;
}
__device__ __forceinline__ void async_wait() {}
#endif

// ---------------------------------------------------------------------------
// Stage 1: Conv1 28->20, 1->256 ch, 9x9 VALID, bias+ReLU.
// Output CHANNEL-LAST f16: h_t[((b*20+iy)*20+ix)*256 + c]  (coalesced stores,
// and makes the PrimaryCaps implicit-GEMM K dimension contiguous).
// ---------------------------------------------------------------------------
__global__ void conv1_relu_f16(const float* __restrict__ img,
                               const float* __restrict__ W1,
                               const float* __restrict__ b1,
                               _Float16* __restrict__ ht) {
  int o = blockIdx.x * blockDim.x + threadIdx.x;
  if (o >= 128 * 400 * 256) return;
  int ch = o & 255; int t = o >> 8;          // pixel index = (b*20+oy)*20+ox
  int ox = t % 20; t /= 20;
  int oy = t % 20; int b = t / 20;
  const float* ip = img + (b * 28 + oy) * 28 + ox;
  const float* wp = W1 + ch * 81;
  float acc = b1[ch];
#pragma unroll
  for (int ky = 0; ky < 9; ++ky)
#pragma unroll
    for (int kx = 0; kx < 9; ++kx)
      acc += ip[ky * 28 + kx] * wp[ky * 9 + kx];
  ht[o] = (_Float16)fmaxf(acc, 0.f);
}

// ---------------------------------------------------------------------------
// Wp (O=256, I=256, 9, 9) f32 -> f16 with K reordered: out[n][r*256+c],
// i.e. (N, K) row-major with K = r*256 + c  (r = ky*9+kx).
// ---------------------------------------------------------------------------
__global__ void cvt_wp_f16(const float* __restrict__ in, _Float16* __restrict__ out) {
  int i = blockIdx.x * blockDim.x + threadIdx.x;
  if (i >= 5308416) return;
  int n = i / 20736, rem = i % 20736;
  int r = rem >> 8, c = rem & 255;
  out[i] = (_Float16)in[n * 20736 + c * 81 + r];
}

// f32 (K,N) -> f16 (N,K) transpose-convert for decoder weights
__global__ void cvt_f16_T(const float* __restrict__ in, _Float16* __restrict__ out,
                          int K, int N) {
  int i = blockIdx.x * blockDim.x + threadIdx.x;
  if (i < K * N) {
    int n = i / K, k = i % K;
    out[i] = (_Float16)in[k * N + n];
  }
}

__device__ __forceinline__ v16h make_a(const v8h lo, const v8h hi) {
  v16h a;
#pragma unroll
  for (int i = 0; i < 8; ++i) { a[i] = lo[i]; a[i + 8] = hi[i]; }
  return a;
}

// ---------------------------------------------------------------------------
// Stage 2: PrimaryCaps conv as implicit GEMM via WMMA.
//   M = 4608 pixels, N = 256 out-ch, K = 81*256 = 20736 (k = r*256 + c).
//   Block: 256 thr = 8 waves (2M x 4N), block tile 64M x 128N.
//   Wave tile 32x32 -> 4 wmma / K-step. A tile (64x32) double-buffered in LDS,
//   streamed with async b128 copies (one per thread). B frags are contiguous
//   v16h loads straight from Wp_t (N,K) rows.
// ---------------------------------------------------------------------------
__launch_bounds__(256)
__global__ void primary_caps_gemm(const _Float16* __restrict__ ht,
                                  const _Float16* __restrict__ Wp,
                                  const float* __restrict__ bp,
                                  float* __restrict__ u_raw) {
  __shared__ __align__(32) _Float16 ldsA[2][64 * 32];
  const int tid  = threadIdx.x;
  const int wave = tid >> 5, lane = tid & 31;
  const int wm = wave >> 2, wn = wave & 3;         // 2 x 4 wave grid
  const int half = lane >> 4, col = lane & 15;
  const int m0 = (blockIdx.x % 72) * 64;
  const int n0 = (blockIdx.x / 72) * 128;

  // staging role: this thread copies 8 halves (16B) of A-tile row mi, seg
  const int mi  = tid >> 2;
  const int seg = (tid & 3) * 8;
  const int ms  = m0 + mi;
  const int bb  = ms / 36, pix = ms % 36;
  const int base0 = bb * 400 + (pix / 6) * 40 + (pix % 6) * 2;  // pixel row base

  const _Float16* brow0 = Wp + (size_t)(n0 + wn * 32 + col) * 20736;
  const _Float16* brow1 = Wp + (size_t)(n0 + wn * 32 + 16 + col) * 20736;

  auto stage = [&](int k0, int buf) {
    const int r = k0 >> 8, c0 = k0 & 255;
    const int ky = r / 9, kx = r % 9;
    const _Float16* src = ht + (size_t)(base0 + ky * 20 + kx) * 256 + c0 + seg;
    async_copy16(&ldsA[buf][mi * 32 + seg], src);
  };

  v8f acc00 = {}, acc01 = {}, acc10 = {}, acc11 = {};
  stage(0, 0);
  async_wait();
  __syncthreads();
  int cur = 0;
  const int rowA0 = wm * 32 + col, rowA1 = rowA0 + 16;
  for (int k0 = 0; k0 < 20736; k0 += 32) {
    if (k0 + 32 < 20736) stage(k0 + 32, cur ^ 1);   // prefetch next tile
    __builtin_prefetch(brow0 + k0 + 512, 0, 1);     // stream Wp ahead (L2)
    __builtin_prefetch(brow1 + k0 + 512, 0, 1);
    // A fragments (rows rowA0 / rowA1): K = 16h + e (contiguous)
    const _Float16* la0 = &ldsA[cur][rowA0 * 32 + 16 * half];
    const _Float16* la1 = &ldsA[cur][rowA1 * 32 + 16 * half];
    v16h a0 = make_a(*(const v8h*)la0, *(const v8h*)(la0 + 8));
    v16h a1 = make_a(*(const v8h*)la1, *(const v8h*)(la1 + 8));
    // B fragments: lane = column, K = k0 + 16*half + e (contiguous)
    const v16h bf0 = *(const v16h*)(brow0 + k0 + 16 * half);
    const v16h bf1 = *(const v16h*)(brow1 + k0 + 16 * half);
    acc00 = __builtin_amdgcn_wmma_f32_16x16x32_f16(false, a0, false, bf0, (short)0, acc00, false, false);
    acc01 = __builtin_amdgcn_wmma_f32_16x16x32_f16(false, a0, false, bf1, (short)0, acc01, false, false);
    acc10 = __builtin_amdgcn_wmma_f32_16x16x32_f16(false, a1, false, bf0, (short)0, acc10, false, false);
    acc11 = __builtin_amdgcn_wmma_f32_16x16x32_f16(false, a1, false, bf1, (short)0, acc11, false, false);
    async_wait();
    __syncthreads();
    cur ^= 1;
  }
  // epilogue: D lane col = n, VGPR i -> M = i + 8*half; scatter to u layout
  auto scatter = [&](const v8f& a, int mbase, int n) {
    const int d = n >> 5, cc = n & 31;
    const float bias = bp[n];
#pragma unroll
    for (int i = 0; i < 8; ++i) {
      int m = mbase + i + 8 * half;
      int b = m / 36, p = m % 36;
      u_raw[((b * 1152) + cc * 36 + p) * 8 + d] = a[i] + bias;
    }
  };
  const int nA = n0 + wn * 32 + col, nB = nA + 16;
  scatter(acc00, m0 + wm * 32,      nA);
  scatter(acc01, m0 + wm * 32,      nB);
  scatter(acc10, m0 + wm * 32 + 16, nA);
  scatter(acc11, m0 + wm * 32 + 16, nB);
}

// ---------------------------------------------------------------------------
// Stage 3: squash u over capsule dim (8)
// ---------------------------------------------------------------------------
__global__ void squash_u(const float* __restrict__ u_raw, float* __restrict__ u_sq) {
  int i = blockIdx.x * blockDim.x + threadIdx.x;
  if (i >= 128 * 1152) return;
  const float* x = u_raw + i * 8;
  float s = 0.f;
#pragma unroll
  for (int d = 0; d < 8; ++d) s += x[d] * x[d];
  float scale = (s / (1.f + s)) * rsqrtf(s);
  float* o = u_sq + i * 8;
#pragma unroll
  for (int d = 0; d < 8; ++d) o[d] = x[d] * scale;
}

// ---------------------------------------------------------------------------
// Stage 4: dynamic routing, 3 iterations. One block per (class c, batch b).
// b_ij kept as per-j scalar (reference b_ij is constant across e); u_hat
// recomputed on the fly from LDS-resident u[b] and L2-resident Wdig.
// ---------------------------------------------------------------------------
__launch_bounds__(256)
__global__ void routing_kernel(const float* __restrict__ u_sq,
                               const float* __restrict__ Wdig,
                               float* __restrict__ caps_out) {
  __shared__ float u_sh[1152 * 8];
  __shared__ float b_sh[1152];
  __shared__ float red[256 * 16];
  __shared__ float sv[16];
  __shared__ float misc[4];
  const int c = blockIdx.x / 128;
  const int b = blockIdx.x % 128;
  const int tid = threadIdx.x;
  for (int i = tid; i < 1152 * 8; i += 256) u_sh[i] = u_sq[b * 1152 * 8 + i];
  for (int j = tid; j < 1152; j += 256) b_sh[j] = 0.f;
  __syncthreads();
  const float* Wc = Wdig + (size_t)c * 1152 * 128;
  float vloc[16];
  for (int it = 0; it < 3; ++it) {
    float lmax = -1e30f;
    for (int j = tid; j < 1152; j += 256) lmax = fmaxf(lmax, b_sh[j]);
    red[tid] = lmax;
    __syncthreads();
    if (tid == 0) {
      float m = -1e30f;
      for (int t = 0; t < 256; ++t) m = fmaxf(m, red[t]);
      misc[0] = m;
    }
    __syncthreads();
    const float bmax = misc[0];
    float lsum = 0.f;
    for (int j = tid; j < 1152; j += 256) lsum += expf(b_sh[j] - bmax);
    red[tid] = lsum;
    __syncthreads();
    if (tid == 0) {
      float s = 0.f;
      for (int t = 0; t < 256; ++t) s += red[t];
      misc[1] = s;
    }
    __syncthreads();
    const float inv_den = 1.f / misc[1];
    float acc[16];
#pragma unroll
    for (int e = 0; e < 16; ++e) acc[e] = 0.f;
    for (int j = tid; j < 1152; j += 256) {
      const float cj = expf(b_sh[j] - bmax) * inv_den;
      const float* wd = Wc + j * 128;
      const float* uu = u_sh + j * 8;
#pragma unroll
      for (int e = 0; e < 16; ++e) {
        float uh = 0.f;
#pragma unroll
        for (int d = 0; d < 8; ++d) uh += uu[d] * wd[d * 16 + e];
        acc[e] += cj * uh;
      }
    }
#pragma unroll
    for (int e = 0; e < 16; ++e) red[tid * 16 + e] = acc[e];
    __syncthreads();
    if (tid < 16) {
      float s = 0.f;
      for (int t = 0; t < 256; ++t) s += red[t * 16 + tid];
      sv[tid] = s;
    }
    __syncthreads();
    if (tid == 0) {
      float sq = 0.f;
      for (int e = 0; e < 16; ++e) sq += sv[e] * sv[e];
      misc[2] = (sq / (1.f + sq)) * rsqrtf(sq);
    }
    __syncthreads();
    const float scale = misc[2];
#pragma unroll
    for (int e = 0; e < 16; ++e) vloc[e] = sv[e] * scale;
    if (it < 2) {
      __syncthreads();
      for (int j = tid; j < 1152; j += 256) {
        const float* wd = Wc + j * 128;
        const float* uu = u_sh + j * 8;
        float dot = 0.f;
#pragma unroll
        for (int e = 0; e < 16; ++e) {
          float uh = 0.f;
#pragma unroll
          for (int d = 0; d < 8; ++d) uh += uu[d] * wd[d * 16 + e];
          dot += uh * vloc[e];
        }
        b_sh[j] += dot;
      }
      __syncthreads();
    }
  }
  if (tid < 16) caps_out[((b * 10) + c) * 16 + tid] = vloc[tid];
}

// ---------------------------------------------------------------------------
// Stage 5: class mask -> one-hot y, masked flat (B,160) f16 for decoder
// ---------------------------------------------------------------------------
__global__ void mask_kernel(const float* __restrict__ caps,
                            float* __restrict__ y_out,
                            _Float16* __restrict__ flat) {
  const int b = blockIdx.x;
  const int tid = threadIdx.x;   // 32
  __shared__ float cls[10];
  __shared__ int best;
  if (tid < 10) {
    float s = 0.f;
    const float* cp = caps + (b * 10 + tid) * 16;
#pragma unroll
    for (int e = 0; e < 16; ++e) s += cp[e] * cp[e];
    cls[tid] = s;   // sqrt & softmax are monotone -> same argmax
  }
  __syncthreads();
  if (tid == 0) {
    int bi = 0; float bv = cls[0];
    for (int c = 1; c < 10; ++c) if (cls[c] > bv) { bv = cls[c]; bi = c; }
    best = bi;
  }
  __syncthreads();
  if (tid < 10) y_out[b * 10 + tid] = (tid == best) ? 1.f : 0.f;
  for (int i = tid; i < 160; i += 32) {
    int c = i >> 4;
    flat[b * 160 + i] = (c == best) ? (_Float16)caps[(b * 10 + c) * 16 + (i & 15)]
                                    : (_Float16)0.f;
  }
}

// ---------------------------------------------------------------------------
// Stage 6: decoder dense layers via WMMA. A (M=128,K) f16 row-major,
// Bt (N,K) f16 pre-transposed, per-wave 16x16 tile, fused bias+act.
// mode 0: ReLU -> f16 out; mode 1: sigmoid -> f32 out.
// ---------------------------------------------------------------------------
__launch_bounds__(128)
__global__ void dense_wmma(const _Float16* __restrict__ A,
                           const _Float16* __restrict__ Bt,
                           const float* __restrict__ bias,
                           _Float16* __restrict__ out_h,
                           float* __restrict__ out_f,
                           int K, int N, int mode) {
  const int ntiles = N >> 4;
  const int ngrp = (ntiles + 3) >> 2;
  const int mt = blockIdx.x / ngrp;
  const int ng = blockIdx.x % ngrp;
  const int wave = threadIdx.x >> 5, lane = threadIdx.x & 31;
  const int nt = ng * 4 + wave;
  if (nt >= ntiles) return;               // wave-uniform: EXEC stays all-ones
  const int half = lane >> 4, col = lane & 15;
  const _Float16* arow = A + (size_t)(mt * 16 + col) * K;
  const int n = nt * 16 + col;
  const _Float16* brow = Bt + (size_t)n * K;
  v8f acc = {};
  for (int k0 = 0; k0 < K; k0 += 32) {
    v16h a = make_a(*(const v8h*)(arow + k0 + 16 * half),
                    *(const v8h*)(arow + k0 + 8 + 16 * half));
    const v16h bf = *(const v16h*)(brow + k0 + 16 * half);
    acc = __builtin_amdgcn_wmma_f32_16x16x32_f16(false, a, false, bf,
                                                 (short)0, acc, false, false);
  }
  const float bs = bias[n];
#pragma unroll
  for (int i = 0; i < 8; ++i) {
    int mm = mt * 16 + i + 8 * half;
    float v = acc[i] + bs;
    if (mode == 0) {
      out_h[(size_t)mm * N + n] = (_Float16)fmaxf(v, 0.f);
    } else {
      out_f[(size_t)mm * N + n] = 1.f / (1.f + expf(-v));
    }
  }
}

// ---------------------------------------------------------------------------
extern "C" void kernel_launch(void* const* d_in, const int* in_sizes, int n_in,
                              void* d_out, int out_size, void* d_ws, size_t ws_size,
                              hipStream_t stream) {
  const float* images = (const float*)d_in[0];
  const float* W1  = (const float*)d_in[1];
  const float* b1  = (const float*)d_in[2];
  const float* Wp  = (const float*)d_in[3];
  const float* bp  = (const float*)d_in[4];
  const float* Wdig= (const float*)d_in[5];
  const float* Wd1 = (const float*)d_in[6];
  const float* bd1 = (const float*)d_in[7];
  const float* Wd2 = (const float*)d_in[8];
  const float* bd2 = (const float*)d_in[9];
  const float* Wd3 = (const float*)d_in[10];
  const float* bd3 = (const float*)d_in[11];

  float* out  = (float*)d_out;
  float* caps = out;                       // (128,10,16)  = 20480
  float* rec  = out + 20480;               // (128,784)    = 100352
  float* yout = out + 20480 + 100352;      // (128,10)     = 1280

  char* ws = (char*)d_ws;
  size_t off = 0;
  auto alloc = [&](size_t bytes) -> void* {
    void* p = ws + off;
    off = (off + bytes + 255) & ~(size_t)255;
    return p;
  };
  _Float16* ht     = (_Float16*)alloc(13107200ull * 2);   // conv1 out, channel-last
  _Float16* Wp_t   = (_Float16*)alloc(5308416ull * 2);    // Wp (N, r*256+c) f16
  float*    u_raw  = (float*)   alloc(1179648ull * 4);    // (B,1152,8)
  float*    u_sq   = (float*)   alloc(1179648ull * 4);
  _Float16* Wd1t   = (_Float16*)alloc(512ull  * 160  * 2);
  _Float16* Wd2t   = (_Float16*)alloc(1024ull * 512  * 2);
  _Float16* Wd3t   = (_Float16*)alloc(784ull  * 1024 * 2);
  _Float16* flat   = (_Float16*)alloc(128ull * 160  * 2);
  _Float16* h1     = (_Float16*)alloc(128ull * 512  * 2);
  _Float16* h2     = (_Float16*)alloc(128ull * 1024 * 2);

  conv1_relu_f16<<<51200, 256, 0, stream>>>(images, W1, b1, ht);
  cvt_wp_f16<<<(5308416 + 255) / 256, 256, 0, stream>>>(Wp, Wp_t);
  cvt_f16_T<<<(512 * 160  + 255) / 256, 256, 0, stream>>>(Wd1, Wd1t, 160, 512);
  cvt_f16_T<<<(1024 * 512 + 255) / 256, 256, 0, stream>>>(Wd2, Wd2t, 512, 1024);
  cvt_f16_T<<<(784 * 1024 + 255) / 256, 256, 0, stream>>>(Wd3, Wd3t, 1024, 784);

  primary_caps_gemm<<<144, 256, 0, stream>>>(ht, Wp_t, bp, u_raw);
  squash_u<<<147456 / 256, 256, 0, stream>>>(u_raw, u_sq);
  routing_kernel<<<1280, 256, 0, stream>>>(u_sq, Wdig, caps);
  mask_kernel<<<128, 32, 0, stream>>>(caps, yout, flat);

  dense_wmma<<<8 * 8,  128, 0, stream>>>(flat, Wd1t, bd1, h1, nullptr, 160, 512, 0);
  dense_wmma<<<8 * 16, 128, 0, stream>>>(h1,  Wd2t, bd2, h2, nullptr, 512, 1024, 0);
  dense_wmma<<<8 * 13, 128, 0, stream>>>(h2,  Wd3t, bd3, nullptr, rec, 1024, 784, 1);
}
